// HybridAttention_27642409517741
// MI455X (gfx1250) — matrix operand order repaired and checked by reading, not compiled, not run on hardware
//
#include <hip/hip_runtime.h>
#include <hip/hip_bf16.h>
#include <math.h>

// ---------------- problem constants (match reference setup_inputs) ----------
#define BATCH   2
#define S_LEN   2048
#define HID     1024
#define NH      16
#define HD      64
#define GAMMA_C 0.5f
#define K_TOP   15            // int(2*log(2048+1)) = 15

typedef __bf16 bf16_t;
typedef __attribute__((ext_vector_type(16))) __bf16 v16bf;
typedef __attribute__((ext_vector_type(8)))  float  v8f;
typedef unsigned int u32x4 __attribute__((ext_vector_type(4)));
typedef int          i32x4 __attribute__((ext_vector_type(4)));
typedef int          i32x8 __attribute__((ext_vector_type(8)));

static __device__ inline v8f vzero8() {
    v8f z;
#pragma unroll
    for (int i = 0; i < 8; ++i) z[i] = 0.0f;
    return z;
}

union V16U { v16bf v; uint4 q[2]; };

// Fragment whose 16 lane-elements are one contiguous 32B run (B-style, transposed tile).
static __device__ inline v16bf ld_frag_contig(const bf16_t* p) {
    V16U u;
    u.q[0] = *(const uint4*)(p);
    u.q[1] = *(const uint4*)(p + 8);
    return u.v;
}
// A-style fragment: one contiguous 32B run at [half*16, +16).
static __device__ inline v16bf ld_frag_a(const bf16_t* row, int half) {
    V16U u;
    u.q[0] = *(const uint4*)(row + half * 16);
    u.q[1] = *(const uint4*)(row + half * 16 + 8);
    return u.v;
}

// ---------------------------------------------------------------------------
// Tensor Data Mover: 2D tile (rowElems x rows, bf16) global -> LDS (row-major).
// D# per CDNA5 ISA 8.3/8.4: group0 {count=1, lds_addr, 57b gaddr, type=2},
// group1 {data_size=1(2B), tensor_dim0/1, tile_dim0/1, tensor_dim0_stride}.
static __device__ inline void tdm_load_2d_bf16(unsigned lds_off, const bf16_t* gptr,
                                               unsigned rowElems, unsigned rows,
                                               unsigned tensorDim0, unsigned tensorDim1,
                                               unsigned long long strideElems) {
    unsigned long long ga = (unsigned long long)(size_t)gptr;
    u32x4 g0;
    g0[0] = 1u;                                               // count=1, user-mode
    g0[1] = lds_off;                                          // LDS byte address
    g0[2] = (unsigned)(ga & 0xffffffffu);                     // global addr lo
    g0[3] = (unsigned)((ga >> 32) & 0x1ffffffu) | (2u << 30); // addr hi | type=2
    i32x8 g1;
    g1[0] = (int)(1u << 16);                                  // data_size=1 -> 2 bytes
    g1[1] = (int)((tensorDim0 & 0xffffu) << 16);              // tensor_dim0[15:0] @ bits63:48
    g1[2] = (int)(((tensorDim0 >> 16) & 0xffffu) | ((tensorDim1 & 0xffffu) << 16));
    g1[3] = (int)(((tensorDim1 >> 16) & 0xffffu) | ((rowElems & 0xffffu) << 16)); // tile_dim0
    g1[4] = (int)(rows & 0xffffu);                            // tile_dim1 (tile_dim2=0)
    g1[5] = (int)(strideElems & 0xffffffffu);                 // tensor_dim0_stride[31:0]
    g1[6] = (int)((strideElems >> 32) & 0xffffu);             // stride[47:32]
    g1[7] = 0;
    i32x4 g2; g2[0] = 0; g2[1] = 0; g2[2] = 0; g2[3] = 0;     // dims 2..3 unused
    i32x4 g3; g3[0] = 0; g3[1] = 0; g3[2] = 0; g3[3] = 0;
#if defined(__clang_major__) && (__clang_major__ >= 23)
    i32x8 g4; g4[0]=0; g4[1]=0; g4[2]=0; g4[3]=0; g4[4]=0; g4[5]=0; g4[6]=0; g4[7]=0;
    __builtin_amdgcn_tensor_load_to_lds(g0, g1, g2, g3, g4, 0);
#else
    __builtin_amdgcn_tensor_load_to_lds(g0, g1, g2, g3, 0);
#endif
}

// ---------------------------------------------------------------------------
// vectorized f32 -> bf16 (n multiple of 4)
__global__ __launch_bounds__(256) void cvt_f32_bf16(const float* __restrict__ in,
                                                    bf16_t* __restrict__ out, int n) {
    int i = (blockIdx.x * 256 + threadIdx.x) * 4;
    if (i >= n) return;
    float4 f = *(const float4*)(in + i);
    union { uint2 q; bf16_t e[4]; } u;
    u.e[0] = (bf16_t)f.x; u.e[1] = (bf16_t)f.y;
    u.e[2] = (bf16_t)f.z; u.e[3] = (bf16_t)f.w;
    *(uint2*)(out + i) = u.q;
}

__global__ __launch_bounds__(256) void zero_f32(float* __restrict__ p, int n) {
    int i = (blockIdx.x * 256 + threadIdx.x) * 4;
    if (i >= n) return;
    float4 z; z.x = 0.f; z.y = 0.f; z.z = 0.f; z.w = 0.f;
    *(float4*)(p + i) = z;
}

// ---------------------------------------------------------------------------
// Band-pass circulant row c[d] from (layer_idx, n_layers) read on device.
__global__ __launch_bounds__(256) void build_filter_row(const int* __restrict__ layer_idx,
                                                        const int* __restrict__ n_layers,
                                                        float* __restrict__ crow) {
    int d = blockIdx.x * 256 + threadIdx.x;
    if (d >= S_LEN) return;
    int li = *layer_idx, nl = *n_layers;
    const int M = S_LEN / 2 + 1;
    const double alpha = 0.5;
    int p, q;
    if (alpha <= 1.0 / (double)nl) {
        p = (int)((double)M * (1.0 - (double)(li + 1) / (double)nl));
        q = (int)((double)p + (double)M / (double)nl);
    } else {
        p = (nl > 1) ? (int)((double)M * (1.0 - alpha) * (1.0 - (double)li / (double)(nl - 1))) : 0;
        q = (int)((double)p + alpha * (double)M);
    }
    p = max(0, min(p, M - 1));
    q = max(p + 1, min(q, M));
    double acc = 0.0;
    const double w0 = 6.283185307179586476925286766559 * (double)d / (double)S_LEN;
    for (int f = p; f < q; ++f) {
        double w = (f == 0 || f == S_LEN / 2) ? 1.0 : 2.0;
        acc += w * cos(w0 * (double)f);
    }
    crow[d] = (float)(acc / (double)S_LEN);
}

__global__ __launch_bounds__(256) void build_filter_mat(const float* __restrict__ crow,
                                                        bf16_t* __restrict__ Cmat) {
    int i = blockIdx.x * 256 + threadIdx.x;
    if (i >= S_LEN * S_LEN) return;
    int t = i >> 11, s = i & (S_LEN - 1);
    Cmat[i] = (bf16_t)crow[(t - s) & (S_LEN - 1)];
}

// ---------------------------------------------------------------------------
// WMMA GEMM: C[M,N] = A[M,K] @ B[K,N] (+bias). Block 256 thr = 8 waves.
// Block tile 128x64, wave tile 16x64, BK=32. Double-buffered pipeline:
// TDM streams the next A tile (row-major) while waves stage the next B tile
// transposed (one b128 global load + 8 b16 LDS scatters per thread) and
// compute WMMAs on the current buffers. One barrier per k-step.
__global__ __launch_bounds__(256)
void wmma_gemm_kernel(const bf16_t* __restrict__ A, const bf16_t* __restrict__ B,
                      const float* __restrict__ bias,
                      float* __restrict__ Cf, bf16_t* __restrict__ Cb,
                      int M, int N, int K) {
    __shared__ bf16_t As[2][128 * 32];   // row-major [m][k]
    __shared__ bf16_t BsT[2][64 * 32];   // transposed [n][k]

    const int tid  = threadIdx.x;
    const int wave = tid >> 5;
    const int lane = tid & 31;
    const int m16  = lane & 15;
    const int half = lane >> 4;
    const int rowBase = blockIdx.y * 128;
    const int colBase = blockIdx.x * 64;

    auto stageA = [&](int buf, int k0) {
        if (tid < 32)
            tdm_load_2d_bf16((unsigned)(size_t)(void*)As[buf],
                             A + (size_t)rowBase * K + k0,
                             32u, 128u, (unsigned)K, (unsigned)(M - rowBase),
                             (unsigned long long)K);
    };
    auto stageB = [&](int buf, int k0) {
        const int r  = tid >> 3;          // 0..31 (k row)
        const int c0 = (tid & 7) * 8;     // 0..56 (col chunk)
        union { uint4 q; bf16_t e[8]; } u;
        u.q = *(const uint4*)&B[(size_t)(k0 + r) * N + (colBase + c0)];
#pragma unroll
        for (int e = 0; e < 8; ++e) BsT[buf][(c0 + e) * 32 + r] = u.e[e];
    };

    v8f acc[4];
#pragma unroll
    for (int t = 0; t < 4; ++t) acc[t] = vzero8();

    stageA(0, 0);
    stageB(0, 0);
    if (tid < 32) __builtin_amdgcn_s_wait_tensorcnt((short)0);
    __syncthreads();

    for (int k0 = 0; k0 < K; k0 += 32) {
        const int cur = (k0 >> 5) & 1, nxt = cur ^ 1;
        if (k0 + 32 < K) { stageA(nxt, k0 + 32); stageB(nxt, k0 + 32); }

        v16bf a = ld_frag_a(&As[cur][(wave * 16 + m16) * 32], half);
#pragma unroll
        for (int nt = 0; nt < 4; ++nt) {
            v16bf bfrag = ld_frag_contig(&BsT[cur][(nt * 16 + m16) * 32 + half * 16]);
            acc[nt] = __builtin_amdgcn_wmma_f32_16x16x32_bf16(
                false, a, false, bfrag, (short)0, acc[nt], false, false);
        }
        if (tid < 32) __builtin_amdgcn_s_wait_tensorcnt((short)0);
        __syncthreads();
    }

#pragma unroll
    for (int nt = 0; nt < 4; ++nt) {
        int col = colBase + nt * 16 + m16;
        float bv = bias ? bias[col] : 0.0f;
#pragma unroll
        for (int r = 0; r < 8; ++r) {
            int row = rowBase + wave * 16 + half * 8 + r;
            float v = acc[nt][r] + bv;
            if (Cb) Cb[(size_t)row * N + col] = (bf16_t)v;
            else    Cf[(size_t)row * N + col] = v;
        }
    }
}

// ---------------------------------------------------------------------------
// Flash attention per (b,h) + autocorrelation wrap-diagonal binning.
// Double-buffered: TDM streams next K tile while waves stage next V tile
// (transposed) and run score/softmax/PV WMMAs on the current buffers.
__global__ __launch_bounds__(256)
void attn_kernel(const bf16_t* __restrict__ Qf, const bf16_t* __restrict__ Kf,
                 const bf16_t* __restrict__ Vf, const float* __restrict__ mask,
                 float* __restrict__ combined, float* __restrict__ acGlobal) {
    __shared__ bf16_t Kt[2][64 * 64];    // [key][d]
    __shared__ bf16_t VtT[2][64 * 64];   // [d][key]
    __shared__ bf16_t Pt[8][16 * 64];    // per-wave P tile [row][key]
    __shared__ float  acLds[S_LEN];

    const int tid  = threadIdx.x;
    const int wave = tid >> 5;
    const int lane = tid & 31;
    const int m16  = lane & 15;
    const int half = lane >> 4;
    const int bh = blockIdx.y;
    const int b  = bh / NH;
    const int h  = bh % NH;
    const int q0 = blockIdx.x * 128 + wave * 16;

    auto stageKV = [&](int buf, int kt) {
        if (tid < 32)
            tdm_load_2d_bf16((unsigned)(size_t)(void*)Kt[buf],
                             Kf + ((size_t)b * S_LEN + kt) * HID + h * HD,
                             64u, 64u, (unsigned)HID, (unsigned)(S_LEN - kt),
                             (unsigned long long)HID);
#pragma unroll
        for (int cc = 0; cc < 2; ++cc) {
            int cid = tid * 2 + cc;       // 0..511
            int key = cid >> 3;           // 0..63
            int c0  = (cid & 7) * 8;      // 0..56
            union { uint4 q; bf16_t e[8]; } u;
            u.q = *(const uint4*)&Vf[((size_t)b * S_LEN + (kt + key)) * HID + h * HD + c0];
#pragma unroll
            for (int e = 0; e < 8; ++e) VtT[buf][(c0 + e) * 64 + key] = u.e[e];
        }
    };

    for (int i = tid; i < S_LEN; i += 256) acLds[i] = 0.0f;

    // Q fragments once (two 32-wide chunks over head dim), b128 global loads
    v16bf qa[2];
    {
        const bf16_t* qrow = Qf + ((size_t)b * S_LEN + (q0 + m16)) * HID + h * HD;
#pragma unroll
        for (int kc = 0; kc < 2; ++kc) qa[kc] = ld_frag_a(qrow + kc * 32, half);
    }

    float mrun[8], lrun[8];
    v8f ctx[4];
#pragma unroll
    for (int r = 0; r < 8; ++r) { mrun[r] = -1e30f; lrun[r] = 0.0f; }
#pragma unroll
    for (int t = 0; t < 4; ++t) ctx[t] = vzero8();

    stageKV(0, 0);
    if (tid < 32) __builtin_amdgcn_s_wait_tensorcnt((short)0);
    __syncthreads();

    for (int kt = 0; kt < S_LEN; kt += 64) {
        const int cur = (kt >> 6) & 1, nxt = cur ^ 1;
        if (kt + 64 < S_LEN) stageKV(nxt, kt + 64);

        // raw scores S = Qfil @ Kfil^T (16x64 per wave); K frag contiguous in Kt
        float sc[4][8];
#pragma unroll
        for (int nt = 0; nt < 4; ++nt) {
            v8f s = vzero8();
#pragma unroll
            for (int kc = 0; kc < 2; ++kc) {
                v16bf bfrag = ld_frag_contig(&Kt[cur][(nt * 16 + m16) * 64 + kc * 32 + half * 16]);
                s = __builtin_amdgcn_wmma_f32_16x16x32_bf16(
                    false, qa[kc], false, bfrag, (short)0, s, false, false);
            }
            const int keyCol = kt + nt * 16 + m16;
#pragma unroll
            for (int r = 0; r < 8; ++r) {
                int qi = q0 + half * 8 + r;
                float raw = s[r];
                atomicAdd(&acLds[(qi - keyCol) & (S_LEN - 1)], raw * (1.0f / (float)HD));
                sc[nt][r] = raw * 0.125f
                          + mask[((size_t)b * S_LEN + qi) * S_LEN + keyCol];
            }
        }

        // online softmax per query row
#pragma unroll
        for (int r = 0; r < 8; ++r) {
            float mx = fmaxf(fmaxf(sc[0][r], sc[1][r]), fmaxf(sc[2][r], sc[3][r]));
#pragma unroll
            for (int off = 1; off < 16; off <<= 1)
                mx = fmaxf(mx, __shfl_xor(mx, off, 32));
            float mnew = fmaxf(mrun[r], mx);
            float corr = __expf(mrun[r] - mnew);
            mrun[r] = mnew;
            lrun[r] *= corr;
#pragma unroll
            for (int dt = 0; dt < 4; ++dt) ctx[dt][r] *= corr;
            float psum = 0.0f;
#pragma unroll
            for (int nt = 0; nt < 4; ++nt) {
                float pv = __expf(sc[nt][r] - mnew);
                sc[nt][r] = pv;
                psum += pv;
            }
#pragma unroll
            for (int off = 1; off < 16; off <<= 1)
                psum += __shfl_xor(psum, off, 32);
            lrun[r] += psum;
        }

        // P tile -> wave-private LDS slab; re-read as A fragments (b128 pairs)
#pragma unroll
        for (int nt = 0; nt < 4; ++nt)
#pragma unroll
            for (int r = 0; r < 8; ++r)
                Pt[wave][(half * 8 + r) * 64 + nt * 16 + m16] = (bf16_t)sc[nt][r];
        asm volatile("s_wait_dscnt 0" ::: "memory");

        // ctx += P @ V ; V fragment contiguous in VtT
#pragma unroll
        for (int dt = 0; dt < 4; ++dt) {
#pragma unroll
            for (int kc = 0; kc < 2; ++kc) {
                v16bf a = ld_frag_a(&Pt[wave][m16 * 64 + kc * 32], half);
                v16bf bfrag = ld_frag_contig(&VtT[cur][(dt * 16 + m16) * 64 + kc * 32 + half * 16]);
                ctx[dt] = __builtin_amdgcn_wmma_f32_16x16x32_bf16(
                    false, a, false, bfrag, (short)0, ctx[dt], false, false);
            }
        }
        if (tid < 32) __builtin_amdgcn_s_wait_tensorcnt((short)0);
        __syncthreads();
    }

    // epilogue: combined = GAMMA * (ctx / l)
#pragma unroll
    for (int dt = 0; dt < 4; ++dt) {
        int col = h * HD + dt * 16 + m16;
#pragma unroll
        for (int r = 0; r < 8; ++r) {
            int row = q0 + half * 8 + r;
            float v = ctx[dt][r] / fmaxf(lrun[r], 1e-20f);
            combined[((size_t)b * S_LEN + row) * HID + col] = GAMMA_C * v;
        }
    }
    for (int i = tid; i < S_LEN; i += 256)
        atomicAdd(&acGlobal[(size_t)bh * S_LEN + i], acLds[i]);
}

// ---------------------------------------------------------------------------
__global__ __launch_bounds__(256)
void topk_kernel(const float* __restrict__ acGlobal,
                 float* __restrict__ topw, int* __restrict__ topidx) {
    __shared__ float vals[S_LEN];
    __shared__ float rv[256];
    __shared__ int   ri[256];
    __shared__ float selV[K_TOP];
    __shared__ int   selI[K_TOP];
    const int bh = blockIdx.x, tid = threadIdx.x;
    for (int i = tid; i < S_LEN; i += 256) vals[i] = acGlobal[(size_t)bh * S_LEN + i];
    __syncthreads();
    for (int it = 0; it < K_TOP; ++it) {
        float best = -1e30f; int bi = 0;
        for (int i = tid; i < S_LEN; i += 256)
            if (vals[i] > best) { best = vals[i]; bi = i; }
        rv[tid] = best; ri[tid] = bi;
        __syncthreads();
        for (int st = 128; st > 0; st >>= 1) {
            if (tid < st && rv[tid + st] > rv[tid]) { rv[tid] = rv[tid + st]; ri[tid] = ri[tid + st]; }
            __syncthreads();
        }
        if (tid == 0) { selV[it] = rv[0]; selI[it] = ri[0]; vals[ri[0]] = -1e30f; }
        __syncthreads();
    }
    if (tid == 0) {
        float mx = selV[0];
        float e[K_TOP], sum = 0.0f;
        for (int i = 0; i < K_TOP; ++i) { e[i] = __expf(selV[i] - mx); sum += e[i]; }
        for (int i = 0; i < K_TOP; ++i) {
            topw[bh * K_TOP + i]   = e[i] / sum;
            topidx[bh * K_TOP + i] = selI[i];
        }
    }
}

// ---------------------------------------------------------------------------
// combined += (1-GAMMA) * sum_k w_k * V[b, (s+idx_k)%S, c] ; 8 channels/thread
__global__ __launch_bounds__(256)
void freq_kernel(const bf16_t* __restrict__ Vbf, const float* __restrict__ topw,
                 const int* __restrict__ topidx, float* __restrict__ combined) {
    size_t t = (size_t)blockIdx.x * 256 + threadIdx.x;
    const size_t total = (size_t)BATCH * S_LEN * (HID / 8);
    if (t >= total) return;
    int c0 = (int)((t % (HID / 8)) * 8);
    int s  = (int)((t / (HID / 8)) % S_LEN);
    int b  = (int)(t / ((size_t)(HID / 8) * S_LEN));
    int bh = b * NH + (c0 / HD);
    float w[K_TOP]; int ix[K_TOP];
#pragma unroll
    for (int i = 0; i < K_TOP; ++i) {
        w[i]  = topw[bh * K_TOP + i];
        ix[i] = topidx[bh * K_TOP + i];
    }
    float sum[8];
#pragma unroll
    for (int e = 0; e < 8; ++e) sum[e] = 0.0f;
#pragma unroll
    for (int i = 0; i < K_TOP; ++i) {
        int sp = (s + ix[i]) & (S_LEN - 1);
        union { uint4 q; bf16_t e[8]; } u;
        u.q = *(const uint4*)&Vbf[((size_t)b * S_LEN + sp) * HID + c0];
#pragma unroll
        for (int e = 0; e < 8; ++e) sum[e] += w[i] * (float)u.e[e];
    }
    float* dst = &combined[((size_t)b * S_LEN + s) * HID + c0];
#pragma unroll
    for (int e = 0; e < 8; ++e) dst[e] += (1.0f - GAMMA_C) * sum[e];
}

// ---------------------------------------------------------------------------
extern "C" void kernel_launch(void* const* d_in, const int* in_sizes, int n_in,
                              void* d_out, int out_size, void* d_ws, size_t ws_size,
                              hipStream_t stream) {
    (void)in_sizes; (void)n_in; (void)out_size; (void)ws_size;
    const float* X    = (const float*)d_in[0];
    const float* mask = (const float*)d_in[1];
    const float* Wq   = (const float*)d_in[2];
    const float* bq   = (const float*)d_in[3];
    const float* Wk   = (const float*)d_in[4];
    const float* bk   = (const float*)d_in[5];
    const float* Wv   = (const float*)d_in[6];
    const float* bv   = (const float*)d_in[7];
    const float* Wo   = (const float*)d_in[8];
    const float* bo   = (const float*)d_in[9];
    const int* layer_idx = (const int*)d_in[10];
    const int* n_layers  = (const int*)d_in[11];
    float* out = (float*)d_out;

    const size_t NTOK = (size_t)BATCH * S_LEN;                 // 4096
    char* ws = (char*)d_ws;
    size_t off = 0;
    auto carve = [&](size_t bytes) -> void* {
        void* p = ws + off;
        off += (bytes + 255) & ~(size_t)255;
        return p;
    };
    bf16_t* Xbf   = (bf16_t*)carve(NTOK * HID * sizeof(bf16_t));
    bf16_t* Wqbf  = (bf16_t*)carve((size_t)HID * HID * sizeof(bf16_t));
    bf16_t* Wkbf  = (bf16_t*)carve((size_t)HID * HID * sizeof(bf16_t));
    bf16_t* Wvbf  = (bf16_t*)carve((size_t)HID * HID * sizeof(bf16_t));
    bf16_t* Wobf  = (bf16_t*)carve((size_t)HID * HID * sizeof(bf16_t));
    bf16_t* Qbf   = (bf16_t*)carve(NTOK * HID * sizeof(bf16_t));
    bf16_t* Kbf   = (bf16_t*)carve(NTOK * HID * sizeof(bf16_t));
    bf16_t* Vbf   = (bf16_t*)carve(NTOK * HID * sizeof(bf16_t));
    bf16_t* Qfil  = (bf16_t*)carve(NTOK * HID * sizeof(bf16_t));
    bf16_t* Kfil  = (bf16_t*)carve(NTOK * HID * sizeof(bf16_t));
    bf16_t* Vfil  = (bf16_t*)carve(NTOK * HID * sizeof(bf16_t));
    bf16_t* Cmat  = (bf16_t*)carve((size_t)S_LEN * S_LEN * sizeof(bf16_t));
    float*  crow  = (float*)carve(S_LEN * sizeof(float));
    float*  comb  = (float*)carve(NTOK * HID * sizeof(float));
    bf16_t* combB = (bf16_t*)carve(NTOK * HID * sizeof(bf16_t));
    float*  acG   = (float*)carve((size_t)BATCH * NH * S_LEN * sizeof(float));
    float*  topw  = (float*)carve((size_t)BATCH * NH * K_TOP * sizeof(float));
    int*    topi  = (int*)carve((size_t)BATCH * NH * K_TOP * sizeof(int));

    const int nX = (int)(NTOK * HID), nW = HID * HID;
    cvt_f32_bf16<<<(nX / 4 + 255) / 256, 256, 0, stream>>>(X, Xbf, nX);
    cvt_f32_bf16<<<(nW / 4 + 255) / 256, 256, 0, stream>>>(Wq, Wqbf, nW);
    cvt_f32_bf16<<<(nW / 4 + 255) / 256, 256, 0, stream>>>(Wk, Wkbf, nW);
    cvt_f32_bf16<<<(nW / 4 + 255) / 256, 256, 0, stream>>>(Wv, Wvbf, nW);
    cvt_f32_bf16<<<(nW / 4 + 255) / 256, 256, 0, stream>>>(Wo, Wobf, nW);
    {
        const int nA = BATCH * NH * S_LEN;
        zero_f32<<<(nA / 4 + 255) / 256, 256, 0, stream>>>(acG, nA);
    }

    build_filter_row<<<(S_LEN + 255) / 256, 256, 0, stream>>>(layer_idx, n_layers, crow);
    build_filter_mat<<<(S_LEN * S_LEN + 255) / 256, 256, 0, stream>>>(crow, Cmat);

    // Q/K/V projections (bf16 out, f32 bias)
    dim3 gProj(HID / 64, (unsigned)(NTOK / 128));
    wmma_gemm_kernel<<<gProj, 256, 0, stream>>>(Xbf, Wqbf, bq, nullptr, Qbf, (int)NTOK, HID, HID);
    wmma_gemm_kernel<<<gProj, 256, 0, stream>>>(Xbf, Wkbf, bk, nullptr, Kbf, (int)NTOK, HID, HID);
    wmma_gemm_kernel<<<gProj, 256, 0, stream>>>(Xbf, Wvbf, bv, nullptr, Vbf, (int)NTOK, HID, HID);

    // Band-pass filter along S as circulant GEMM, per batch
    dim3 gFil(HID / 64, S_LEN / 128);
    for (int b = 0; b < BATCH; ++b) {
        const size_t o = (size_t)b * S_LEN * HID;
        wmma_gemm_kernel<<<gFil, 256, 0, stream>>>(Cmat, Qbf + o, nullptr, nullptr, Qfil + o, S_LEN, HID, S_LEN);
        wmma_gemm_kernel<<<gFil, 256, 0, stream>>>(Cmat, Kbf + o, nullptr, nullptr, Kfil + o, S_LEN, HID, S_LEN);
        wmma_gemm_kernel<<<gFil, 256, 0, stream>>>(Cmat, Vbf + o, nullptr, nullptr, Vfil + o, S_LEN, HID, S_LEN);
    }

    // Flash attention + autocorrelation binning
    attn_kernel<<<dim3(S_LEN / 128, BATCH * NH), 256, 0, stream>>>(Qfil, Kfil, Vfil, mask, comb, acG);

    // top-k and frequency branch
    topk_kernel<<<BATCH * NH, 256, 0, stream>>>(acG, topw, topi);
    freq_kernel<<<(unsigned)((NTOK * (HID / 8) + 255) / 256), 256, 0, stream>>>(Vbf, topw, topi, comb);

    // output projection
    cvt_f32_bf16<<<(nX / 4 + 255) / 256, 256, 0, stream>>>(comb, combB, nX);
    wmma_gemm_kernel<<<gProj, 256, 0, stream>>>(combB, Wobf, bo, out, nullptr, (int)NTOK, HID, HID);
}
